// MultiHeadSelfAttention_28716151341281
// MI455X (gfx1250) — matrix operand order repaired and checked by reading, not compiled, not run on hardware
//
#include <hip/hip_runtime.h>
#include <hip/hip_bf16.h>

// MHA + RoPE for MI455X (gfx1250, wave32, WMMA bf16 + double-buffered TDM).
// B=2, T=2048, D=1024, H=16, dk=64.

#define B_   2
#define T_   2048
#define D_   1024
#define H_   16
#define DK_  64

typedef __attribute__((ext_vector_type(16))) __bf16 v16bf;
typedef __attribute__((ext_vector_type(8)))  __bf16 v8bf;
typedef __attribute__((ext_vector_type(8)))  float  v8f;
typedef __attribute__((ext_vector_type(4)))  unsigned int u32x4;
typedef __attribute__((ext_vector_type(8)))  int i32x8;
typedef __attribute__((ext_vector_type(4)))  int i32x4;

#if __has_builtin(__builtin_amdgcn_tensor_load_to_lds)
#define HAVE_TDM 1
#else
#define HAVE_TDM 0
#endif

// ---------------------------------------------------------------------------
// Tensor Data Mover: 2-D tile (rows x row_elems bf16) global -> LDS.
// D# built per CDNA5 ISA ch.8: group0 = {flags, lds_addr, global_addr, type=2},
// group1 = {data_size=2B, tensor_dim0/1, tile_dim0/1, tensor_dim0_stride}.
// ---------------------------------------------------------------------------
#if HAVE_TDM
__device__ __forceinline__ void tdm_load_2d(unsigned lds_off, const void* gptr,
                                            unsigned rows, unsigned row_elems,
                                            unsigned stride_elems) {
  unsigned long long ga = (unsigned long long)(size_t)gptr;
  u32x4 g0;
  g0[0] = 1u;                                            // count=1 user desc
  g0[1] = lds_off;                                       // lds_addr (bytes)
  g0[2] = (unsigned)(ga & 0xFFFFFFFFu);                  // global_addr[31:0]
  g0[3] = (unsigned)((ga >> 32) & 0x1FFFFFFu) | (2u << 30); // addr[56:32], type=2
  i32x8 g1;
  g1[0] = (int)(1u << 16);                               // data_size=1 (2 bytes)
  g1[1] = (int)((row_elems & 0xFFFFu) << 16);            // tensor_dim0[15:0]
  g1[2] = (int)(((row_elems >> 16) & 0xFFFFu) | ((rows & 0xFFFFu) << 16));
  g1[3] = (int)(((rows >> 16) & 0xFFFFu) | ((row_elems & 0xFFFFu) << 16)); // tile_dim0
  g1[4] = (int)(rows & 0xFFFFu);                         // tile_dim1 (tile_dim2=0)
  g1[5] = (int)stride_elems;                             // tensor_dim0_stride[31:0]
  g1[6] = 0;
  g1[7] = 0;
  i32x4 z4 = {0, 0, 0, 0};
#if defined(__clang_major__) && (__clang_major__ >= 23)
  i32x8 z8 = {0, 0, 0, 0, 0, 0, 0, 0};
  __builtin_amdgcn_tensor_load_to_lds(g0, g1, z4, z4, z8, 0);
#else
  __builtin_amdgcn_tensor_load_to_lds(g0, g1, z4, z4, 0);
#endif
}
#endif

// ---------------------------------------------------------------------------
// WMMA helpers (CDNA5 16x16x32 bf16 -> f32 accumulate)
// ---------------------------------------------------------------------------
__device__ __forceinline__ v8f wmma_bf16(v16bf a, v16bf b, v8f c) {
  return __builtin_amdgcn_wmma_f32_16x16x32_bf16(false, a, false, b,
                                                 (short)0, c, false, false);
}

// A-fragment (16x32 bf16): lanes 0-15 hold row M=ln K{0-15}; lanes 16-31
// hold K{16-31}. One 32B load at element offset half*16.
__device__ __forceinline__ v16bf load_a_frag(const __bf16* row_k0, int half) {
  return *(const v16bf*)(row_k0 + half * 16);
}

// B-fragment (32x16 bf16): lane holds column N=ln, 16 contiguous K at
// offset half*16. Source stored (N, K) row-major.
__device__ __forceinline__ v16bf load_b_frag(const __bf16* coln_k0, int half) {
  return *(const v16bf*)(coln_k0 + half * 16);
}

// ---------------------------------------------------------------------------
// fp32 -> bf16 cast (8 elements / thread)
// ---------------------------------------------------------------------------
__global__ void k_f32_to_bf16(const float* __restrict__ src,
                              __bf16* __restrict__ dst, int n) {
  int i = (blockIdx.x * blockDim.x + threadIdx.x) * 8;
  if (i >= n) return;
  float4 a = *(const float4*)(src + i);
  float4 b = *(const float4*)(src + i + 4);
  v8bf o;
  o[0] = (__bf16)a.x; o[1] = (__bf16)a.y; o[2] = (__bf16)a.z; o[3] = (__bf16)a.w;
  o[4] = (__bf16)b.x; o[5] = (__bf16)b.y; o[6] = (__bf16)b.z; o[7] = (__bf16)b.w;
  *(v8bf*)(dst + i) = o;
}

// ---------------------------------------------------------------------------
// QKV projection: qkv[m,o] = sum_d x[m,d] * Wqkv[o,d]  (M=4096, N=3072, K=1024)
// 64 blocks x 256 threads (8 waves). A 64x1024 bf16 panel (128 KB) staged in
// dynamic LDS via one TDM descriptor; wave (mt,nh) owns 16 rows x 1536 cols.
// Epilogue applies RoPE (lane-pair shfl) and scatters q/k (B,H,T,dk), vT (B,H,dk,T).
// ---------------------------------------------------------------------------
__global__ __launch_bounds__(256)
void k_qkv_gemm_rope(const __bf16* __restrict__ xb, const __bf16* __restrict__ wb,
                     const int* __restrict__ tp,
                     __bf16* __restrict__ qb, __bf16* __restrict__ kb,
                     __bf16* __restrict__ vtb) {
  extern __shared__ __attribute__((aligned(32))) __bf16 xs[];  // 64*1024 bf16
  const int tid   = threadIdx.x;
  const int mbase = blockIdx.x * 64;

#if HAVE_TDM
  if (tid < 32) {   // one TDM descriptor moves the whole 128 KB panel
    tdm_load_2d((unsigned)(size_t)&xs[0], xb + (size_t)mbase * 1024,
                64, 1024, 1024);
    __builtin_amdgcn_s_wait_tensorcnt(0);
  }
#else
  {
    const uint4* src = (const uint4*)(xb + (size_t)mbase * 1024);
    uint4* dst = (uint4*)xs;
#pragma unroll
    for (int j = 0; j < 32; ++j) dst[tid + j * 256] = src[tid + j * 256];
  }
#endif
  __syncthreads();

  const int lane = tid & 31, wave = tid >> 5;
  const int ln = lane & 15, half = lane >> 4;
  const int mt = wave >> 1, nh = wave & 1;
  const __bf16* arow = xs + (size_t)(mt * 16 + ln) * 1024;

  float pos[8]; int bidx[8], trow[8];
#pragma unroll
  for (int r = 0; r < 8; ++r) {
    int m = mbase + mt * 16 + r + 8 * half;    // C-layout row for vgpr r
    bidx[r] = m >> 11;
    trow[r] = m & (T_ - 1);
    pos[r]  = (float)tp[m];
  }

  for (int nt = 0; nt < 96; ++nt) {
    const int o = nh * 1536 + nt * 16 + ln;    // this lane's output column
    const __bf16* wrow = wb + (size_t)o * 1024;
    __builtin_prefetch(wb + (size_t)(o + 16) * 1024, 0, 1);
    v8f acc = {};
#pragma unroll 4
    for (int kc = 0; kc < 32; ++kc) {
      const int k0 = kc * 32;
      acc = wmma_bf16(load_a_frag(arow + k0, half),
                      load_b_frag(wrow + k0, half), acc);
    }

    const int sec = o >> 10;                   // 0=Q 1=K 2=V
    const int oo  = o & 1023;
    const int h = oo >> 6, d = oo & 63;
    if (sec == 2) {                            // V: store transposed (B,H,dk,T)
#pragma unroll
      for (int r = 0; r < 8; ++r)
        vtb[(((size_t)bidx[r] * H_ + h) * DK_ + d) * T_ + trow[r]] = (__bf16)acc[r];
    } else {                                   // Q/K: fused RoPE
      const float inv_freq = __expf(-0.2878231366f * (float)(d >> 1));
      __bf16* base = (sec == 0) ? qb : kb;
#pragma unroll
      for (int r = 0; r < 8; ++r) {
        float v  = acc[r];
        float pv = __shfl_xor(v, 1, 32);       // even/odd feature in lane^1
        float s, c;
        __sincosf(pos[r] * inv_freq, &s, &c);
        float out = (d & 1) ? (v * c + pv * s) : (v * c - pv * s);
        base[(((size_t)bidx[r] * H_ + h) * T_ + trow[r]) * DK_ + d] = (__bf16)out;
      }
    }
  }
}

// ---------------------------------------------------------------------------
// Flash-style causal attention. Block = 4 waves = 64 queries of one (b,h).
// 32-key K/V chunks are TDM-staged into double-buffered LDS: chunk i+1 is
// issued before chunk i is consumed; TENSORcnt is in-order per wave, so
// s_wait_tensorcnt<=2 guarantees chunk i's two loads have landed while the
// next two are still in flight. End-of-iteration barrier makes the refill
// target (last read in the previous iteration) safe.
// ---------------------------------------------------------------------------
__global__ __launch_bounds__(128)
void k_attn(const __bf16* __restrict__ qb, const __bf16* __restrict__ kb,
            const __bf16* __restrict__ vtb, __bf16* __restrict__ cb) {
  __shared__ __attribute__((aligned(32))) __bf16 Ks[2][32 * 64];   // (key, dk)
  __shared__ __attribute__((aligned(32))) __bf16 Vs[2][64 * 32];   // (dk, key)
  __shared__ __attribute__((aligned(32))) __bf16 Pt[4 * 16 * 32];  // per-wave P

  const int tid = threadIdx.x;
  const int lane = tid & 31, wave = tid >> 5;
  const int ln = lane & 15, half = lane >> 4;
  const int bh = blockIdx.x, b = bh >> 4, h = bh & 15;
  const int qblk  = blockIdx.y * 64;
  const int qbase = qblk + wave * 16;
  const int nchunks = qblk / 32 + 2;           // ceil((qblk+63+1)/32)

  const __bf16* qrow  = qb  + (((size_t)bh) * T_ + qbase) * DK_;
  const __bf16* krow  = kb  + ((size_t)bh) * T_ * DK_;
  const __bf16* vbase = vtb + ((size_t)bh) * DK_ * T_;
  __bf16* ptw = Pt + wave * (16 * 32);

  v16bf aq0 = load_a_frag(qrow + (size_t)ln * DK_ +  0, half);
  v16bf aq1 = load_a_frag(qrow + (size_t)ln * DK_ + 32, half);

  v8f o0 = {}, o1 = {}, o2 = {}, o3 = {};
  float mi[8], li[8];
#pragma unroll
  for (int r = 0; r < 8; ++r) { mi[r] = -1e30f; li[r] = 0.f; }

#if HAVE_TDM
  if (tid < 32) {                              // prologue: chunk 0 -> buffer 0
    tdm_load_2d((unsigned)(size_t)&Ks[0][0], krow,  32, 64, 64);
    tdm_load_2d((unsigned)(size_t)&Vs[0][0], vbase, 64, 32, T_);
  }
#endif

  for (int ci = 0; ci < nchunks; ++ci) {
    const int jb  = ci * 32;
    const int cur = ci & 1;
#if HAVE_TDM
    if (tid < 32) {
      if (ci + 1 < nchunks) {                  // prefetch chunk i+1 -> other buf
        tdm_load_2d((unsigned)(size_t)&Ks[cur ^ 1][0],
                    krow + (size_t)(jb + 32) * DK_, 32, 64, 64);
        tdm_load_2d((unsigned)(size_t)&Vs[cur ^ 1][0],
                    vbase + jb + 32, 64, 32, T_);
        __builtin_amdgcn_s_wait_tensorcnt(2);  // chunk i done; i+1 in flight
      } else {
        __builtin_amdgcn_s_wait_tensorcnt(0);
      }
    }
#else
    {
      const uint4* ksrc = (const uint4*)(krow + (size_t)jb * DK_);
      uint4* kdst = (uint4*)&Ks[cur][0];
#pragma unroll
      for (int j = 0; j < 2; ++j) kdst[tid + j * 128] = ksrc[tid + j * 128];
#pragma unroll
      for (int j = 0; j < 2; ++j) {
        int idx = tid + j * 128;               // 256 uint4 total
        int d = idx >> 2, c = idx & 3;
        ((uint4*)&Vs[cur][0])[idx] =
            *((const uint4*)(vbase + (size_t)d * T_ + jb) + c);
      }
    }
#endif
    __syncthreads();

    // ---- S = Q K^T for two 16-wide key tiles -----------------------------
    const __bf16* k0r = &Ks[cur][0] + (size_t)ln * DK_;
    const __bf16* k1r = &Ks[cur][0] + (size_t)(16 + ln) * DK_;
    v8f s0 = {}, s1 = {};
    s0 = wmma_bf16(aq0, load_b_frag(k0r,      half), s0);
    s0 = wmma_bf16(aq1, load_b_frag(k0r + 32, half), s0);
    s1 = wmma_bf16(aq0, load_b_frag(k1r,      half), s1);
    s1 = wmma_bf16(aq1, load_b_frag(k1r + 32, half), s1);

#pragma unroll
    for (int r = 0; r < 8; ++r) {              // scale + causal mask
      int qi = qbase + r + 8 * half;
      s0[r] = (jb      + ln <= qi) ? s0[r] * 0.125f : -1e30f;
      s1[r] = (jb + 16 + ln <= qi) ? s1[r] * 0.125f : -1e30f;
    }

    float alpha[8];
#pragma unroll
    for (int r = 0; r < 8; ++r) {              // online softmax
      float mx = fmaxf(s0[r], s1[r]);
#pragma unroll
      for (int off = 1; off < 16; off <<= 1) mx = fmaxf(mx, __shfl_xor(mx, off, 16));
      float mnew = fmaxf(mi[r], mx);
      alpha[r] = __expf(mi[r] - mnew);
      float p0 = __expf(s0[r] - mnew), p1 = __expf(s1[r] - mnew);
      float rs = p0 + p1;
#pragma unroll
      for (int off = 1; off < 16; off <<= 1) rs += __shfl_xor(rs, off, 16);
      li[r] = li[r] * alpha[r] + rs;
      mi[r] = mnew;
      ptw[(r + 8 * half) * 32 + ln]      = (__bf16)p0;   // C-layout -> LDS
      ptw[(r + 8 * half) * 32 + 16 + ln] = (__bf16)p1;
    }

    // ---- O = O*alpha + P V (per-wave P; same-wave LDS ops are in-order) --
    v16bf ap = load_a_frag(ptw + ln * 32, half);
#pragma unroll
    for (int r = 0; r < 8; ++r) {
      o0[r] *= alpha[r]; o1[r] *= alpha[r]; o2[r] *= alpha[r]; o3[r] *= alpha[r];
    }
    const __bf16* vs = &Vs[cur][0];
    o0 = wmma_bf16(ap, load_b_frag(vs + ( 0 + ln) * 32, half), o0);
    o1 = wmma_bf16(ap, load_b_frag(vs + (16 + ln) * 32, half), o1);
    o2 = wmma_bf16(ap, load_b_frag(vs + (32 + ln) * 32, half), o2);
    o3 = wmma_bf16(ap, load_b_frag(vs + (48 + ln) * 32, half), o3);
    __syncthreads();                           // refill target safe next iter
  }

  // ---- normalize + write concat (B,T,D) bf16 ------------------------------
#pragma unroll
  for (int r = 0; r < 8; ++r) {
    float inv = 1.0f / li[r];
    int t = qbase + r + 8 * half;
    __bf16* dst = cb + ((size_t)b * T_ + t) * D_ + h * DK_;
    dst[ 0 + ln] = (__bf16)(o0[r] * inv);
    dst[16 + ln] = (__bf16)(o1[r] * inv);
    dst[32 + ln] = (__bf16)(o2[r] * inv);
    dst[48 + ln] = (__bf16)(o3[r] * inv);
  }
}

// ---------------------------------------------------------------------------
// Output projection: out[m,o] = sum_d concat[m,d] * Wout[o,d]
// (M=4096, N=1024, K=1024). Same TDM-staged 64-row panel structure.
// ---------------------------------------------------------------------------
__global__ __launch_bounds__(256)
void k_out_gemm(const __bf16* __restrict__ cb, const __bf16* __restrict__ wb,
                float* __restrict__ out) {
  extern __shared__ __attribute__((aligned(32))) __bf16 xs[];  // 64*1024 bf16
  const int tid = threadIdx.x;
  const int mbase = blockIdx.x * 64;

#if HAVE_TDM
  if (tid < 32) {
    tdm_load_2d((unsigned)(size_t)&xs[0], cb + (size_t)mbase * 1024,
                64, 1024, 1024);
    __builtin_amdgcn_s_wait_tensorcnt(0);
  }
#else
  {
    const uint4* src = (const uint4*)(cb + (size_t)mbase * 1024);
    uint4* dst = (uint4*)xs;
#pragma unroll
    for (int j = 0; j < 32; ++j) dst[tid + j * 256] = src[tid + j * 256];
  }
#endif
  __syncthreads();

  const int lane = tid & 31, wave = tid >> 5;
  const int ln = lane & 15, half = lane >> 4;
  const int mt = wave >> 1, nh = wave & 1;
  const __bf16* arow = xs + (size_t)(mt * 16 + ln) * 1024;

  for (int nt = 0; nt < 32; ++nt) {
    const int o = nh * 512 + nt * 16 + ln;
    const __bf16* wrow = wb + (size_t)o * 1024;
    __builtin_prefetch(wb + (size_t)(o + 16) * 1024, 0, 1);
    v8f acc = {};
#pragma unroll 4
    for (int kc = 0; kc < 32; ++kc) {
      acc = wmma_bf16(load_a_frag(arow + kc * 32, half),
                      load_b_frag(wrow + kc * 32, half), acc);
    }
#pragma unroll
    for (int r = 0; r < 8; ++r)
      out[(size_t)(mbase + mt * 16 + r + 8 * half) * 1024 + o] = acc[r];
  }
}

// ---------------------------------------------------------------------------
// Launcher. Workspace layout (needs 48 MB):
//   [0,8M)   x  bf16        [8M,14M)  Wqkv bf16     [14M,16M) Wout bf16
//   [16,24M) q  bf16 roped  [24,32M)  k bf16 roped  [32,40M)  v^T bf16
//   [40,48M) attn concat bf16
// ---------------------------------------------------------------------------
extern "C" void kernel_launch(void* const* d_in, const int* in_sizes, int n_in,
                              void* d_out, int out_size, void* d_ws, size_t ws_size,
                              hipStream_t stream) {
  (void)in_sizes; (void)n_in; (void)out_size; (void)ws_size;
  const float* x    = (const float*)d_in[0];
  const int*   tp   = (const int*)d_in[1];
  const float* wqkv = (const float*)d_in[2];
  const float* wout = (const float*)d_in[3];
  float* out = (float*)d_out;

  char* ws = (char*)d_ws;
  __bf16* xb  = (__bf16*)(ws);
  __bf16* wqb = (__bf16*)(ws + ( 8ull << 20));
  __bf16* wob = (__bf16*)(ws + (14ull << 20));
  __bf16* qb  = (__bf16*)(ws + (16ull << 20));
  __bf16* kb  = (__bf16*)(ws + (24ull << 20));
  __bf16* vtb = (__bf16*)(ws + (32ull << 20));
  __bf16* cb  = (__bf16*)(ws + (40ull << 20));

  k_f32_to_bf16<<<2048, 256, 0, stream>>>(x,    xb,  4194304);
  k_f32_to_bf16<<<1536, 256, 0, stream>>>(wqkv, wqb, 3145728);
  k_f32_to_bf16<<< 512, 256, 0, stream>>>(wout, wob, 1048576);

  const size_t panel_lds = 64 * 1024 * sizeof(__bf16);   // 128 KB dynamic LDS
  k_qkv_gemm_rope<<<64, 256, panel_lds, stream>>>(xb, wqb, tp, qb, kb, vtb);
  k_attn<<<dim3(B_ * H_, T_ / 64), 128, 0, stream>>>(qb, kb, vtb, cb);
  k_out_gemm<<<64, 256, panel_lds, stream>>>(cb, wob, out);
}
